// SymplecticLayer_32134945308676
// MI455X (gfx1250) — matrix operand order repaired and checked by reading, not compiled, not run on hardware
//
#include <hip/hip_runtime.h>
#include <hip/hip_bf16.h>

// ---- types ----
typedef __attribute__((ext_vector_type(16))) __bf16 v16bf;
typedef __attribute__((ext_vector_type(8)))  __bf16 v8bf;
typedef __attribute__((ext_vector_type(8)))  float  v8f;

#define DT_STEP 0.01f
#define HID 128
#define DIMZ 8
#define ROWS_PER_WAVE 16
#define WAVES 8
#define ROWS_PER_BLOCK (ROWS_PER_WAVE * WAVES)   // 128
#define NSTEPS 2

// Branch-free tanh on the hardware transcendental pipe:
// tanh(x) = 1 - 2/(1 + exp2(x * 2*log2(e))).  TRANS ops co-execute with WMMA.
__device__ __forceinline__ float fast_tanh(float x) {
  float e = __builtin_amdgcn_exp2f(x * 2.88539008177792681f);  // exp(2x)
  float r = __builtin_amdgcn_rcpf(1.0f + e);
  return 1.0f - 2.0f * r;
}

// A-matrix 16x32 bf16 fragment from row-major [16][HID] bf16 buffer.
// ISA layout: lane<16 holds row=lane, K = {kc..kc+7} U {kc+16..kc+23};
//             lane>=16 holds row=lane-16, K = {kc+8..kc+15} U {kc+24..kc+31}.
__device__ __forceinline__ v16bf load_a_frag(const __bf16* __restrict__ buf,
                                             int lane, int kc) {
  int row = lane & 15;
  int kb  = kc + ((lane & 16) ? 8 : 0);
  const __bf16* p = buf + row * HID + kb;
  v8bf lo = *(const v8bf*)(p);
  v8bf hi = *(const v8bf*)(p + 16);
  v16bf a;
#pragma unroll
  for (int i = 0; i < 8; ++i) { a[i] = lo[i]; a[i + 8] = hi[i]; }
  return a;
}

// B-matrix 32x16 bf16 fragment from a [N][ldk] buffer (K contiguous per column).
// ISA layout: lanes 0-15 = col n, K {kc..kc+7} U {kc+16..kc+23}; lanes 16-31 = col n, K {kc+8..kc+15} U {kc+24..kc+31}.
__device__ __forceinline__ v16bf load_b_frag(const __bf16* __restrict__ buf,
                                             int ldk, int lane, int nb, int kc) {
  int n  = nb + (lane & 15);
  int kb = kc + ((lane & 16) ? 8 : 0);
  const __bf16* p = buf + n * ldk + kb;
  v8bf lo = *(const v8bf*)(p);
  v8bf hi = *(const v8bf*)(p + 16);
  v16bf b;
#pragma unroll
  for (int i = 0; i < 8; ++i) { b[i] = lo[i]; b[i + 8] = hi[i]; }
  return b;
}

__device__ __forceinline__ v8f wmma_bf16(v16bf a, v16bf b, v8f c) {
  return __builtin_amdgcn_wmma_f32_16x16x32_bf16(false, a, false, b,
                                                 (short)0, c, false, false);
}

__global__ __launch_bounds__(256)
void hnn_leapfrog_kernel(const float* __restrict__ z,
                         const float* __restrict__ W1,
                         const float* __restrict__ b1,
                         const float* __restrict__ W2,
                         const float* __restrict__ b2,
                         const float* __restrict__ W3,
                         float* __restrict__ out) {
  // LDS: weights in both orientations for contiguous B-fragment loads.
  __shared__ __bf16 sW1cm[HID * 32];        // [n][k]: k<8 = W1[k][n], k==8 = b1[n] (bias folded)
  __shared__ __bf16 sW1T [16 * HID];        // [n][k] = W1[n][k], n<8 valid (dz B)
  __shared__ __bf16 sW2cm[HID * HID];       // [n][k] = W2[k][n]      (forward L2 B)
  __shared__ __bf16 sW2rm[HID * HID];       // [n][k] = W2[n][k]      (backward B = W2^T colmajor)
  __shared__ float  sB2[HID];
  __shared__ float  sW3[HID];
  __shared__ __bf16 sStage[WAVES][ROWS_PER_WAVE * HID];  // per-wave layout staging
  __shared__ float  sQP[ROWS_PER_BLOCK * DIMZ];          // q (0..3), p (4..7)

  const int tid  = threadIdx.x;
  const int lane = tid & 31;
  const int wave = tid >> 5;
  const long long blockBase = (long long)blockIdx.x * (ROWS_PER_BLOCK * DIMZ);

  // ---- async copy of the z tile into LDS (overlaps weight conversion) ----
  // 256 threads x 16B = 4KB = exactly ROWS_PER_BLOCK*DIMZ floats.
  {
    unsigned ldsaddr = (unsigned)(unsigned long long)(&sQP[tid * 4]);
    unsigned voff    = (unsigned)(tid * 16);
    const float* zsrc = z + blockBase;
    asm volatile("global_load_async_to_lds_b128 %0, %1, %2 offset:0"
                 :
                 : "v"(ldsaddr), "v"(voff), "s"(zsrc)
                 : "memory");
  }

  // Warm L2 for the weights (global_prefetch_b8).
  __builtin_prefetch(W2 + tid * 64, 0, 2);

  // ---- stage weights ----
  for (int i = tid; i < HID * 32; i += 256) {
    int n = i >> 5, k = i & 31;
    __bf16 v = (__bf16)0.0f;
    if (k < DIMZ)       v = (__bf16)W1[k * HID + n];
    else if (k == DIMZ) v = (__bf16)b1[n];          // bias row (A supplies 1.0 at K=8)
    sW1cm[i] = v;
  }
  for (int i = tid; i < 16 * HID; i += 256) {
    int n = i >> 7, k = i & 127;
    sW1T[i] = (n < DIMZ) ? (__bf16)W1[n * HID + k] : (__bf16)0.0f;
  }
  for (int i = tid; i < HID * HID; i += 256) {
    int n = i >> 7, k = i & 127;
    sW2cm[i] = (__bf16)W2[k * HID + n];
    sW2rm[i] = (__bf16)W2[i];
  }
  for (int i = tid; i < HID; i += 256) {
    sB2[i] = b2[i]; sW3[i] = W3[i];
  }

  // z tile must have landed before any wave reads sQP.
  asm volatile("s_wait_asynccnt 0" ::: "memory");
  __syncthreads();

  float*  qp    = &sQP[wave * ROWS_PER_WAVE * DIMZ];
  __bf16* stage = sStage[wave];
  const int col   = lane & 15;
  const int rhalf = (lane & 16) ? 8 : 0;

  // 4 gradient evaluations: (full, half) x NSTEPS
#pragma unroll 1
  for (int it = 0; it < 2 * NSTEPS; ++it) {
    const bool full = ((it & 1) == 0);

    // ---- layer 1: pre1 = [z,1] @ [W1;b1]  (K=9 zero-padded to 32) ----
    v16bf az;
#pragma unroll
    for (int i = 0; i < 16; ++i) az[i] = (__bf16)0.0f;
    if (lane < 16) {
      const float* zp = qp + (lane & 15) * DIMZ;
#pragma unroll
      for (int i = 0; i < DIMZ; ++i) az[i] = (__bf16)zp[i];
    } else {
      az[0] = (__bf16)1.0f;   // K=8 -> multiplies the bias row of sW1cm
    }
    float d1[8][8];  // 1 - tanh^2(pre1), C layout, per column tile
#pragma unroll
    for (int j = 0; j < 8; ++j) {
      const int nb = j * 16;
      v8f acc;
#pragma unroll
      for (int v = 0; v < 8; ++v) acc[v] = 0.0f;   // bias folded into GEMM
      v16bf bw = load_b_frag(sW1cm, 32, lane, nb, 0);
      acc = wmma_bf16(az, bw, acc);
#pragma unroll
      for (int v = 0; v < 8; ++v) {
        float t = fast_tanh(acc[v]);
        d1[j][v] = 1.0f - t * t;
        stage[(rhalf + v) * HID + nb + col] = (__bf16)t;   // h1 -> staging
      }
    }

    // ---- layer 2: pre2 = h1 @ W2 + b2 ; u2 = W3 * (1 - tanh^2(pre2)) ----
    v16bf aH[4];
#pragma unroll
    for (int c = 0; c < 4; ++c) aH[c] = load_a_frag(stage, lane, c * 32);
#pragma unroll
    for (int j = 0; j < 8; ++j) {
      const int nb = j * 16;
      const float bias = sB2[nb + col];
      v8f acc;
#pragma unroll
      for (int v = 0; v < 8; ++v) acc[v] = bias;
#pragma unroll
      for (int c = 0; c < 4; ++c) {
        v16bf bw = load_b_frag(sW2cm, HID, lane, nb, c * 32);
        acc = wmma_bf16(aH[c], bw, acc);
      }
      const float w3v = sW3[nb + col];
#pragma unroll
      for (int v = 0; v < 8; ++v) {
        float t  = fast_tanh(acc[v]);
        float u2 = w3v * (1.0f - t * t);
        stage[(rhalf + v) * HID + nb + col] = (__bf16)u2;  // u2 -> staging
      }
    }

    // ---- backward through layer 2: u1 = (u2 @ W2^T) * d1 ----
    v16bf aU[4];
#pragma unroll
    for (int c = 0; c < 4; ++c) aU[c] = load_a_frag(stage, lane, c * 32);
#pragma unroll
    for (int j = 0; j < 8; ++j) {
      const int nb = j * 16;
      v8f acc;
#pragma unroll
      for (int v = 0; v < 8; ++v) acc[v] = 0.0f;
#pragma unroll
      for (int c = 0; c < 4; ++c) {
        v16bf bw = load_b_frag(sW2rm, HID, lane, nb, c * 32);
        acc = wmma_bf16(aU[c], bw, acc);
      }
#pragma unroll
      for (int v = 0; v < 8; ++v) {
        float u1 = acc[v] * d1[j][v];
        stage[(rhalf + v) * HID + nb + col] = (__bf16)u1;  // u1 -> staging
      }
    }

    // ---- dz = u1 @ W1^T (one 16x16 tile, cols 0..7 valid) ----
    v16bf aG[4];
#pragma unroll
    for (int c = 0; c < 4; ++c) aG[c] = load_a_frag(stage, lane, c * 32);
    v8f g;
#pragma unroll
    for (int v = 0; v < 8; ++v) g[v] = 0.0f;
#pragma unroll
    for (int c = 0; c < 4; ++c) {
      v16bf bw = load_b_frag(sW1T, HID, lane, 0, c * 32);
      g = wmma_bf16(aG[c], bw, g);
    }

    // ---- leapfrog update in LDS ----
    if (col < 4) {
      // dHdq (cols 0..3) -> p kick (half step)
#pragma unroll
      for (int v = 0; v < 8; ++v)
        qp[(rhalf + v) * DIMZ + 4 + col] -= 0.5f * DT_STEP * g[v];
    } else if (col < 8) {
      // dHdp (cols 4..7) -> q drift (only on the "full" eval)
      if (full) {
#pragma unroll
        for (int v = 0; v < 8; ++v)
          qp[(rhalf + v) * DIMZ + (col - 4)] += DT_STEP * g[v];
      }
    }
  }

  __syncthreads();
  for (int i = tid; i < ROWS_PER_BLOCK * DIMZ; i += 256) {
    out[blockBase + i] = sQP[i];
  }
}

extern "C" void kernel_launch(void* const* d_in, const int* in_sizes, int n_in,
                              void* d_out, int out_size, void* d_ws, size_t ws_size,
                              hipStream_t stream) {
  const float* z  = (const float*)d_in[0];
  const float* W1 = (const float*)d_in[1];
  const float* b1 = (const float*)d_in[2];
  const float* W2 = (const float*)d_in[3];
  const float* b2 = (const float*)d_in[4];
  const float* W3 = (const float*)d_in[5];
  // d_in[6] = b3: gradient of H w.r.t. z is independent of b3.
  float* out = (float*)d_out;

  const int batch  = in_sizes[0] / DIMZ;          // 524288
  const int blocks = batch / ROWS_PER_BLOCK;      // 4096
  hnn_leapfrog_kernel<<<blocks, 256, 0, stream>>>(z, W1, b1, W2, b2, W3, out);
}
